// NN_chain_82454782149148
// MI455X (gfx1250) — matrix operand order repaired and checked by reading, not compiled, not run on hardware
//
#include <hip/hip_runtime.h>

// ---------------------------------------------------------------------------
// 51 chained 1x1 convs (64->64) over 401408 pixels, fp32 in/out.
// Compute-bound (168 GFLOP vs 206 MB => ~814 FLOP/B) -> f16 WMMA with fp32
// accumulation: V_WMMA_F32_16X16X32_F16 (8192 MACs/instr vs 1024 for f32 K=4).
//
// Per layer: H_out(64 x Np) = W(64x64) * H_in(64 x Np) + b.
// Each wave owns a 64-chan x 32-pixel tile (2 pixel groups of 16):
//   8 fp32 accum tiles, 16 WMMA/layer, A fragments register-reused across the
//   two pixel groups, bias folded into the WMMA C-operand init.
// Weights: pre-converted to f16 once (L2-resident), then double-buffered into
// LDS per layer via GLOBAL_LOAD_ASYNC_TO_LDS_B128 (ASYNCcnt path) overlapped
// with compute; one s_wait_asynccnt + barrier per layer.
//
// Inter-layer D-layout -> B-layout transpose goes through a per-wave private
// LDS tile. The two pixel-group values a lane holds for one channel are
// stored interleaved (half index 2*ln + g), so each pair is one
// v_cvt_pk_f16_f32 + ds_store_b32. loadB reads back the same 2*ln + g slot, so the
// pixel labeling is unchanged by each of the 50 stagings before the last layer,
// so the net permutation at the
// final store is the identity and the output mapping is unchanged.
// ---------------------------------------------------------------------------

typedef __attribute__((ext_vector_type(16))) _Float16 v16h;
typedef __attribute__((ext_vector_type(8)))  _Float16 v8h;
typedef __attribute__((ext_vector_type(2)))  _Float16 v2h;
typedef __attribute__((ext_vector_type(8)))  float    v8f;
typedef __attribute__((ext_vector_type(4)))  float    v4f;

#define N_LAYERS 51
#define CH       64
#define HW       12544              // 112*112
#define NPIX     (32 * HW)          // 401408
#define PPW      32                 // pixels per wave (2 groups of 16)
#define WPB      8                  // waves per block
#define THREADS  (WPB * 32)

// ---- pre-pass: convert weights fp32 -> f16 once (418 KB, L2-resident) ----
__global__ __launch_bounds__(256)
void cvt_weights_f16(const float* __restrict__ W, _Float16* __restrict__ Wh, int n)
{
    int i = blockIdx.x * blockDim.x + threadIdx.x;
    if (i < n) Wh[i] = (_Float16)W[i];
}

__device__ __forceinline__ v8f wmma_f16(v16h a, v16h b, v8f c)
{
    // (neg_a, A, neg_b, B, c_mod, C, reuse_a, reuse_b)
    return __builtin_amdgcn_wmma_f32_16x16x32_f16(false, a, false, b,
                                                  (short)0, c, false, false);
}

// Low 32 bits of a generic pointer to LDS == LDS byte offset on amdgcn.
__device__ __forceinline__ unsigned lds_off32(const void* p)
{
    return (unsigned)(uintptr_t)p;
}

// Stage one 8 KB f16 weight layer into LDS via the CDNA5 async-to-LDS path.
// Each of the 256 threads moves 32 bytes (2x b128). Tracked by ASYNCcnt.
__device__ __forceinline__ void async_stage_w(const _Float16* __restrict__ src,
                                              unsigned lds_base_byte)
{
    unsigned voff = (unsigned)threadIdx.x * 32u;
    asm volatile(
        "global_load_async_to_lds_b128 %0, %1, %2\n\t"
        "global_load_async_to_lds_b128 %0, %1, %2 offset:16"
        :
        : "v"(lds_base_byte + voff), "v"(voff), "s"(src)
        : "memory");
}

__device__ __forceinline__ void wait_async_zero()
{
    asm volatile("s_wait_asynccnt 0x0" ::: "memory");
}

__global__ __launch_bounds__(THREADS)
void conv_chain_kernel(const float* __restrict__ x,
                       const _Float16* __restrict__ Wh,
                       const float* __restrict__ bias,
                       float* __restrict__ out)
{
    __shared__ __align__(64) _Float16 sW[2][CH * CH];    // 2 x 8 KB weights (f16)
    __shared__ __align__(64) float    sB[2][CH];         // 2 x 256 B bias (f32)
    __shared__ __align__(64) _Float16 sH[WPB][CH * PPW]; // per-wave h staging (32 KB)

    const int tid  = threadIdx.x;
    const int wave = tid >> 5;
    const int lane = tid & 31;
    const int hi   = lane >> 4;       // lane-half select
    const int ln   = lane & 15;

    // 32 consecutive pixels per wave; 12544 % 32 == 0 so no batch crossing.
    const long   tile  = (long)blockIdx.x * WPB + wave;
    const long   p0    = tile * PPW;
    const long   bimg  = p0 / HW;
    const long   hw0   = p0 - bimg * HW;
    const size_t gbase = (size_t)bimg * CH * HW + (size_t)hw0;  // + c*HW per chan

    // ---- stage layer-0 weights/bias into buffer 0 (async path) ----
    async_stage_w(Wh, lds_off32(&sW[0][0]));
    if (tid < CH) sB[0][tid] = bias[tid];
    wait_async_zero();
    __syncthreads();

    // A fragment (16x32 f16, documented layout): lane-half K interleave.
    auto loadA = [&](int buf, int m, int k) -> v16h {
        const _Float16* w = sW[buf] + (m * 16 + ln) * CH + k * 32 + hi * 8;
        v8h lo = *(const v8h*)w;          // K = {0..7} / {8..15}
        v8h hp = *(const v8h*)(w + 16);   // K = {16..23} / {24..31}
        v16h a;
#pragma unroll
        for (int j = 0; j < 8; ++j) { a[j] = lo[j]; a[j + 8] = hp[j]; }
        return a;
    };

    // B fragment (32x16 f16): lane = pixel column (slot 2*ln+g), K in table order.
    auto loadB = [&](int k, int g) -> v16h {
        const _Float16* h = sH[wave] + (k * 32 + hi * 8) * PPW + ln * 2 + g;
        v16h b;
#pragma unroll
        for (int j = 0; j < 8; ++j) { b[j] = h[j * PPW]; b[j + 8] = h[(j + 16) * PPW]; }
        return b;
    };

    // Bias in D layout: element r of tile m is channel m*16 + hi*8 + r.
    auto loadBiasFrag = [&](int buf, int m) -> v8f {
        const float* bp = sB[buf] + m * 16 + hi * 8;
        v4f x0 = *(const v4f*)bp;
        v4f x1 = *(const v4f*)(bp + 4);
        v8f r;
#pragma unroll
        for (int j = 0; j < 4; ++j) { r[j] = x0[j]; r[j + 4] = x1[j]; }
        return r;
    };

    // One layer: 16 WMMAs; bias pre-loaded into C; A reused for both groups.
    auto computeLayer = [&](int buf, const v16h bf[2][2], v8f acc0[4], v8f acc1[4]) {
#pragma unroll
        for (int m = 0; m < 4; ++m) {
            v8f binit = loadBiasFrag(buf, m);   // D = W*h + bias via C operand
            acc0[m] = binit;
            acc1[m] = binit;
        }
#pragma unroll
        for (int k = 0; k < 2; ++k) {
#pragma unroll
            for (int m = 0; m < 4; ++m) {
                v16h a = loadA(buf, m, k);
                acc0[m] = wmma_f16(a, bf[0][k], acc0[m]);
                acc1[m] = wmma_f16(a, bf[1][k], acc1[m]);
            }
        }
    };

    // D layout -> chan-major f16 staging, pixel slots interleaved (2*ln + g)
    // so each lane emits one packed cvt + one ds_store_b32 per (m, r).
    auto storeH = [&](v8f acc0[4], v8f acc1[4]) {
#pragma unroll
        for (int m = 0; m < 4; ++m) {
#pragma unroll
            for (int r = 0; r < 8; ++r) {
                int chan = m * 16 + hi * 8 + r;
                v2h p;
                p[0] = (_Float16)acc0[m][r];
                p[1] = (_Float16)acc1[m][r];
                *(v2h*)(sH[wave] + chan * PPW + ln * 2) = p;
            }
        }
    };

    // After 50 stagings the slot permutation is identity: slot = g*16 + ln.
    auto storeOut = [&](v8f acc0[4], v8f acc1[4]) {
#pragma unroll
        for (int m = 0; m < 4; ++m) {
#pragma unroll
            for (int r = 0; r < 8; ++r) {
                int chan = m * 16 + hi * 8 + r;
                out[gbase + (size_t)chan * HW + 0  + ln] = acc0[m][r];
                out[gbase + (size_t)chan * HW + 16 + ln] = acc1[m][r];
            }
        }
    };

    v8f acc0[4], acc1[4];
    int cur = 0;

    // ---- layer 0 (B operands straight from global x) ----
    {
        // kick off layer-1 weights while we compute
        async_stage_w(Wh + (size_t)1 * CH * CH, lds_off32(&sW[1][0]));
        float bn = 0.f;
        if (tid < CH) bn = bias[1 * CH + tid];

        v16h bf[2][2];
#pragma unroll
        for (int g = 0; g < 2; ++g) {
#pragma unroll
            for (int k = 0; k < 2; ++k) {
                const float* px = x + gbase + (size_t)(k * 32 + hi * 8) * HW + g * 16 + ln;
                v16h b;
#pragma unroll
                for (int j = 0; j < 8; ++j) {
                    b[j]     = (_Float16)px[(size_t)j * HW];
                    b[j + 8] = (_Float16)px[(size_t)(j + 16) * HW];
                }
                bf[g][k] = b;
            }
        }

        computeLayer(0, bf, acc0, acc1);
        storeH(acc0, acc1);
        if (tid < CH) sB[1][tid] = bn;
        wait_async_zero();          // layer-1 weights landed in sW[1]
        __syncthreads();
        cur = 1;
    }

    // ---- layers 1..50: one barrier per layer, async double-buffered weights ----
    for (int l = 1; l < N_LAYERS; ++l) {
        const bool last = (l == N_LAYERS - 1);
        float bn = 0.f;
        if (!last) {
            // safe: every wave here has passed barrier(l-1), so all reads of
            // sW[cur^1] (done during layer l-1) have completed.
            async_stage_w(Wh + (size_t)(l + 1) * CH * CH, lds_off32(&sW[cur ^ 1][0]));
            if (tid < CH) bn = bias[(l + 1) * CH + tid];
        }

        v16h bf[2][2];
#pragma unroll
        for (int g = 0; g < 2; ++g)
#pragma unroll
            for (int k = 0; k < 2; ++k)
                bf[g][k] = loadB(k, g);

        computeLayer(cur, bf, acc0, acc1);

        if (last) {
            storeOut(acc0, acc1);        // fp32 D (bias already in C) -> global
        } else {
            storeH(acc0, acc1);
            if (tid < CH) sB[cur ^ 1][tid] = bn;
            wait_async_zero();
            __syncthreads();
            cur ^= 1;
        }
    }
}

extern "C" void kernel_launch(void* const* d_in, const int* in_sizes, int n_in,
                              void* d_out, int out_size, void* d_ws, size_t ws_size,
                              hipStream_t stream)
{
    (void)in_sizes; (void)n_in; (void)out_size; (void)ws_size;

    const float* x = (const float*)d_in[0];            // (32,64,112,112) f32
    const float* W = (const float*)d_in[1];            // (51,64,64) f32
    const float* b = (const float*)d_in[2];            // (51,64) f32
    float*    outp = (float*)d_out;                    // (32,64,112,112) f32
    _Float16*   Wh = (_Float16*)d_ws;                  // 51*64*64 f16 = 418 KB

    const int nW = N_LAYERS * CH * CH;                 // 208896
    cvt_weights_f16<<<(nW + 255) / 256, 256, 0, stream>>>(W, Wh, nW);

    const int nBlocks = NPIX / (WPB * PPW);            // 1568, exact
    conv_chain_kernel<<<nBlocks, THREADS, 0, stream>>>(x, Wh, b, outp);
}
